// GRUDecoder_30648886624822
// MI455X (gfx1250) — matrix operand order repaired and checked by reading, not compiled, not run on hardware
//
#include <hip/hip_runtime.h>
#include <hip/hip_bf16.h>
#include <stdint.h>

// ---------------- problem constants ----------------
#define E_DIM 256
#define T_LEN 512
#define B_SZ  128
#define S_LEN 64
#define E2    (2 * E_DIM)   // 512
#define E3    (3 * E_DIM)   // 768
#define GW    16            // workgroups cooperating per batch tile
#define MB    16            // batch rows per group (WMMA M)
#define NGRP  (B_SZ / MB)   // 8 groups
#define TC    32            // attention rows staged per TDM chunk
#define NCHUNK (T_LEN / TC) // 16

typedef float    v8f   __attribute__((ext_vector_type(8)));
typedef __bf16   v16bf __attribute__((ext_vector_type(16)));
typedef unsigned u32x4 __attribute__((ext_vector_type(4)));
typedef int      i32x4 __attribute__((ext_vector_type(4)));
typedef int      i32x8 __attribute__((ext_vector_type(8)));

union BfTile { v16bf v; uint4 q[2]; };

// ---------------- workspace layout (bytes) ----------------
#define WS_H      0u                        // f32  [B,E]
#define WS_V      (WS_H     + 131072u)      // f32  [B,E]
#define WS_GX     (WS_V     + 131072u)      // f32  [B,3E]
#define WS_GH     (WS_GX    + 393216u)      // f32  [B,3E]
#define WS_HBF    (WS_GH    + 393216u)      // bf16 [B,E]
#define WS_INPBF  (WS_HBF   + 65536u)       // bf16 [B,2E]
#define WS_WWBF   (WS_INPBF + 131072u)      // bf16 [E,E]
#define WS_WIHBF  (WS_WWBF  + 131072u)      // bf16 [3E,2E]
#define WS_WHHBF  (WS_WIHBF + 786432u)      // bf16 [3E,E]
#define WS_BAR    (WS_WHHBF + 393216u)      // u32  [2*NGRP]

// ---------------- 16x16 bf16 WMMA tile: D += A(16xK) * B(KxN=W^T) ----------------
// A row-major [16, lda] (activation rows), Bm row-major [16 rows of W, ldb] where
// out[m,n] = sum_k A[m,k] * Bm[n,k]  (i.e. Bm holds W rows, GEMM with W^T).
__device__ inline v8f gemm_tile(const __bf16* __restrict__ A, int lda,
                                const __bf16* __restrict__ Bm, int ldb,
                                int K, int lane) {
  v8f c = {0.f, 0.f, 0.f, 0.f, 0.f, 0.f, 0.f, 0.f};
  const int m  = lane & 15;
  const int hi = lane >> 4;
  const __bf16* ar = A  + (size_t)m * lda;
  const __bf16* br = Bm + (size_t)m * ldb;
  for (int k0 = 0; k0 < K; k0 += 32) {
    BfTile a, b;
    // 16-bit A 16x32 layout: lanes0-15 K=[k0..k0+7],[k0+16..k0+23]; lanes16-31 +8
    a.q[0] = *(const uint4*)(ar + k0 + 8 * hi);
    a.q[1] = *(const uint4*)(ar + k0 + 16 + 8 * hi);
    // B 32x16: same K order as A (lanes0-15 K=[k0..k0+7],[k0+16..k0+23]; lanes16-31 +8), N=lane&15
    b.q[0] = *(const uint4*)(br + k0 + 8 * hi);
    b.q[1] = *(const uint4*)(br + k0 + 16 + 8 * hi);
    c = __builtin_amdgcn_wmma_f32_16x16x32_bf16(false, a.v, false, b.v,
                                                (short)0, c, false, false);
  }
  return c;
}

// ---------------- TDM: 2D tile (rows x 256 f32) global -> LDS ----------------
__device__ inline void tdm_load_sig(const float* ldsdst, const float* gsrc, int rows) {
  unsigned lds_off = (unsigned)(unsigned long long)(uintptr_t)ldsdst;
  unsigned long long ga = (unsigned long long)(uintptr_t)gsrc;
  u32x4 g0;
  g0[0] = 1u;                                   // count=1 valid descriptor
  g0[1] = lds_off;                              // lds_addr (bytes)
  g0[2] = (unsigned)ga;                         // global_addr[31:0]
  g0[3] = (unsigned)((ga >> 32) & 0x1FFFFFFull) | (2u << 30); // addr[56:32], type=2
  i32x8 g1;
  g1[0] = (int)(2u << 16);                      // data_size=2 -> 4 bytes
  g1[1] = (int)((unsigned)(E_DIM & 0xFFFF) << 16);          // tensor_dim0 lo16
  g1[2] = (int)((unsigned)(T_LEN & 0xFFFF) << 16);          // dim0 hi | tensor_dim1 lo
  g1[3] = (int)((unsigned)(E_DIM & 0xFFFF) << 16);          // dim1 hi | tile_dim0=256
  g1[4] = (int)(rows & 0xFFFF);                 // tile_dim1 = rows, tile_dim2=0
  g1[5] = E_DIM;                                // tensor_dim0_stride lo32
  g1[6] = 0;
  g1[7] = 0;
  i32x4 z4 = {0, 0, 0, 0};
  i32x8 z8 = {0, 0, 0, 0, 0, 0, 0, 0};
  // clang-23 / therock-10.0 toolchain: 6-arg form (g0, g1, g2, g3, g4, cpol)
  __builtin_amdgcn_tensor_load_to_lds(g0, g1, z4, z4, z8, 0);
}

// ---------------- inter-WG barrier among the 16 WGs of a group ----------------
__device__ inline void group_barrier(unsigned* cnt, unsigned* gen, int g, unsigned target) {
  __syncthreads();
  if (threadIdx.x == 0) {
    __threadfence();
    unsigned prev = __hip_atomic_fetch_add(&cnt[g], 1u, __ATOMIC_ACQ_REL,
                                           __HIP_MEMORY_SCOPE_AGENT);
    if (prev == GW - 1) {
      __hip_atomic_store(&cnt[g], 0u, __ATOMIC_RELAXED, __HIP_MEMORY_SCOPE_AGENT);
      __hip_atomic_fetch_add(&gen[g], 1u, __ATOMIC_RELEASE, __HIP_MEMORY_SCOPE_AGENT);
    }
    while (__hip_atomic_load(&gen[g], __ATOMIC_ACQUIRE, __HIP_MEMORY_SCOPE_AGENT) < target)
      __builtin_amdgcn_s_sleep(2);
    __threadfence();
  }
  __syncthreads();
}

// ---------------- prep: weights -> bf16, init h, reset barriers ----------------
__global__ void gru_prep_kernel(const float* __restrict__ Ww,
                                const float* __restrict__ Wih,
                                const float* __restrict__ Whh,
                                const float* __restrict__ hinit, char* ws) {
  float*   hY    = (float*)(ws + WS_H);
  __bf16*  hbf   = (__bf16*)(ws + WS_HBF);
  __bf16*  wwbf  = (__bf16*)(ws + WS_WWBF);
  __bf16*  wihbf = (__bf16*)(ws + WS_WIHBF);
  __bf16*  whhbf = (__bf16*)(ws + WS_WHHBF);
  unsigned* bar  = (unsigned*)(ws + WS_BAR);
  const int total = E3 * E2;  // largest conversion (393216)
  for (int i = blockIdx.x * blockDim.x + threadIdx.x; i < total;
       i += gridDim.x * blockDim.x) {
    wihbf[i] = (__bf16)Wih[i];
    if (i < E3 * E_DIM)    whhbf[i] = (__bf16)Whh[i];
    if (i < E_DIM * E_DIM) wwbf[i]  = (__bf16)Ww[i];
    if (i < B_SZ * E_DIM) {
      float v = hinit[i & (E_DIM - 1)];
      hY[i] = v;
      hbf[i] = (__bf16)v;
    }
    if (i < 2 * NGRP) bar[i] = 0u;
  }
}

// ---------------- persistent decoder ----------------
__global__ __launch_bounds__(256, 1) void gru_decoder_kernel(
    const float* __restrict__ signal, const float* __restrict__ bases,
    const unsigned char* __restrict__ msk,
    const float* __restrict__ W_b, const float* __restrict__ ln_g,
    const float* __restrict__ ln_b, const float* __restrict__ gbi,
    const float* __restrict__ gbh, float* __restrict__ out, char* ws) {
  float*   hY    = (float*)(ws + WS_H);
  float*   vbuf  = (float*)(ws + WS_V);
  float*   gxw   = (float*)(ws + WS_GX);
  float*   ghw   = (float*)(ws + WS_GH);
  __bf16*  hbf   = (__bf16*)(ws + WS_HBF);
  __bf16*  inpbf = (__bf16*)(ws + WS_INPBF);
  const __bf16* wwbf  = (const __bf16*)(ws + WS_WWBF);
  const __bf16* wihbf = (const __bf16*)(ws + WS_WIHBF);
  const __bf16* whhbf = (const __bf16*)(ws + WS_WHHBF);
  unsigned* barcnt = (unsigned*)(ws + WS_BAR);
  unsigned* bargen = barcnt + NGRP;

  const int wg = blockIdx.x;      // 0..127
  const int g  = wg / GW;         // batch-tile group
  const int wl = wg % GW;         // local id within group
  const int b  = wg;              // attention batch element owned by this WG
  const int b0 = g * MB;
  const int tid  = threadIdx.x;
  const int lane = tid & 31;
  const int wid  = tid >> 5;
  const int hi   = lane >> 4;
  const int m16  = lane & 15;

  __shared__ float sig[2][TC * E_DIM];       // 64 KB double-buffered signal tile
  __shared__ float red_acc[8][E_DIM];        // flash merge accumulators
  __shared__ float red_m[8], red_l[8];
  __shared__ float red_s[8], red_s2[8];

  unsigned bt = 0;

  for (int step = 0; step < S_LEN; ++step) {
    // ---- P1: v = h @ Ww^T + b  (WMMA; this WG owns n-tile wl) ----
    if (wid == 0) {
      const int n0 = wl * 16;
      v8f c = gemm_tile(hbf + (size_t)b0 * E_DIM, E_DIM,
                        wwbf + (size_t)n0 * E_DIM, E_DIM, E_DIM, lane);
      const int n = n0 + m16;
      const float bias = W_b[n];
#pragma unroll
      for (int r = 0; r < 8; ++r)
        vbuf[(size_t)(b0 + r + 8 * hi) * E_DIM + n] = c[r] + bias;
    }
    group_barrier(barcnt, bargen, g, ++bt);

    // ---- P2: flash attention over T for batch element b ----
    float vr[8];
#pragma unroll
    for (int r = 0; r < 8; ++r)
      vr[r] = vbuf[(size_t)b * E_DIM + 8 * lane + r];
    float mrun = -1e30f, lrun = 0.f;
    float acc[8];
#pragma unroll
    for (int r = 0; r < 8; ++r) acc[r] = 0.f;

    if (tid == 0) tdm_load_sig(&sig[0][0], signal + (size_t)b * T_LEN * E_DIM, TC);

    for (int ch = 0; ch < NCHUNK; ++ch) {
      const int buf = ch & 1;
      if (wid == 0) __builtin_amdgcn_s_wait_tensorcnt(0);
      __syncthreads();                       // chunk `ch` resident in sig[buf]
      if (tid == 0 && ch + 1 < NCHUNK)
        tdm_load_sig(&sig[buf ^ 1][0],
                     signal + ((size_t)b * T_LEN + (size_t)(ch + 1) * TC) * E_DIM, TC);
      const int t0 = ch * TC;
#pragma unroll
      for (int j = 0; j < TC / 8; ++j) {     // 4 rows per wave
        const int row = wid * (TC / 8) + j;
        const int t = t0 + row;
        const float* sr = &sig[buf][row * E_DIM + 8 * lane];
        const float4 s0 = *(const float4*)(sr);
        const float4 s1 = *(const float4*)(sr + 4);
        float d = s0.x * vr[0] + s0.y * vr[1] + s0.z * vr[2] + s0.w * vr[3] +
                  s1.x * vr[4] + s1.y * vr[5] + s1.z * vr[6] + s1.w * vr[7];
#pragma unroll
        for (int off = 16; off > 0; off >>= 1) d += __shfl_xor(d, off);
        const float e  = msk[(size_t)b * T_LEN + t] ? -1e30f : d;
        const float mn = fmaxf(mrun, e);
        const float sc = __expf(mrun - mn);
        const float p  = __expf(e - mn);
        lrun = lrun * sc + p;
        acc[0] = acc[0] * sc + p * s0.x;  acc[1] = acc[1] * sc + p * s0.y;
        acc[2] = acc[2] * sc + p * s0.z;  acc[3] = acc[3] * sc + p * s0.w;
        acc[4] = acc[4] * sc + p * s1.x;  acc[5] = acc[5] * sc + p * s1.y;
        acc[6] = acc[6] * sc + p * s1.z;  acc[7] = acc[7] * sc + p * s1.w;
        mrun = mn;
      }
      __syncthreads();                       // done reading sig[buf]
    }
    // merge 8 per-wave flash states
#pragma unroll
    for (int r = 0; r < 8; ++r) red_acc[wid][8 * lane + r] = acc[r];
    if (lane == 0) { red_m[wid] = mrun; red_l[wid] = lrun; }
    __syncthreads();
    float M = red_m[0];
#pragma unroll
    for (int w = 1; w < 8; ++w) M = fmaxf(M, red_m[w]);
    float L = 0.f, cx = 0.f;
#pragma unroll
    for (int w = 0; w < 8; ++w) {
      const float f = __expf(red_m[w] - M);
      L += red_l[w] * f;
      cx += f * red_acc[w][tid];
    }
    const float ctx = cx / L;                // context[b][tid]

    // ---- P3: LayerNorm over concat(bases[b,step,:], ctx) ----
    const float x1 = bases[((size_t)b * S_LEN + step) * E_DIM + tid];
    const float x2 = ctx;
    float s = x1 + x2, s2 = x1 * x1 + x2 * x2;
#pragma unroll
    for (int off = 16; off > 0; off >>= 1) {
      s += __shfl_xor(s, off);
      s2 += __shfl_xor(s2, off);
    }
    if (lane == 0) { red_s[wid] = s; red_s2[wid] = s2; }
    __syncthreads();
    float S1 = 0.f, S2 = 0.f;
#pragma unroll
    for (int w = 0; w < 8; ++w) { S1 += red_s[w]; S2 += red_s2[w]; }
    const float mu  = S1 * (1.f / 512.f);
    const float var = S2 * (1.f / 512.f) - mu * mu;
    const float rs  = rsqrtf(var + 1e-5f);
    inpbf[(size_t)b * E2 + tid]         = (__bf16)((x1 - mu) * rs * ln_g[tid] + ln_b[tid]);
    inpbf[(size_t)b * E2 + E_DIM + tid] = (__bf16)((x2 - mu) * rs * ln_g[E_DIM + tid] +
                                                   ln_b[E_DIM + tid]);
    group_barrier(barcnt, bargen, g, ++bt);

    // ---- P4: GRU GEMMs gx = inp@Wih^T, gh = h@Whh^T (WMMA; 6 jobs on waves 0-5) ----
    if (wid < 3) {
      const int n0 = (wl * 3 + wid) * 16;
      v8f c = gemm_tile(inpbf + (size_t)b0 * E2, E2,
                        wihbf + (size_t)n0 * E2, E2, E2, lane);
      const int n = n0 + m16;
      const float bias = gbi[n];
#pragma unroll
      for (int r = 0; r < 8; ++r)
        gxw[(size_t)(b0 + r + 8 * hi) * E3 + n] = c[r] + bias;
    } else if (wid < 6) {
      const int n0 = (wl * 3 + (wid - 3)) * 16;
      v8f c = gemm_tile(hbf + (size_t)b0 * E_DIM, E_DIM,
                        whhbf + (size_t)n0 * E_DIM, E_DIM, E_DIM, lane);
      const int n = n0 + m16;
      const float bias = gbh[n];
#pragma unroll
      for (int r = 0; r < 8; ++r)
        ghw[(size_t)(b0 + r + 8 * hi) * E3 + n] = c[r] + bias;
    }
    group_barrier(barcnt, bargen, g, ++bt);

    // ---- P5: gates + hidden update for batch element b ----
    {
      const size_t ro = (size_t)b * E3;
      const float xr = gxw[ro + tid];
      const float xz = gxw[ro + E_DIM + tid];
      const float xn = gxw[ro + E2 + tid];
      const float hr = ghw[ro + tid];
      const float hz = ghw[ro + E_DIM + tid];
      const float hn = ghw[ro + E2 + tid];
      const float hold = hY[(size_t)b * E_DIM + tid];
      const float r = 1.f / (1.f + __expf(-(xr + hr)));
      const float z = 1.f / (1.f + __expf(-(xz + hz)));
      const float n = tanhf(xn + r * hn);
      const float hnew = (1.f - z) * n + z * hold;
      hY[(size_t)b * E_DIM + tid] = hnew;
      hbf[(size_t)b * E_DIM + tid] = (__bf16)hnew;
      out[((size_t)b * S_LEN + step) * E_DIM + tid] = hnew;
    }
    group_barrier(barcnt, bargen, g, ++bt);
  }
}

extern "C" void kernel_launch(void* const* d_in, const int* in_sizes, int n_in,
                              void* d_out, int out_size, void* d_ws, size_t ws_size,
                              hipStream_t stream) {
  const float* signal = (const float*)d_in[0];
  const float* bases  = (const float*)d_in[1];
  const unsigned char* msk = (const unsigned char*)d_in[2];
  const float* W_w  = (const float*)d_in[3];
  const float* W_b  = (const float*)d_in[4];
  const float* ln_g = (const float*)d_in[5];
  const float* ln_b = (const float*)d_in[6];
  const float* gwih = (const float*)d_in[7];
  const float* gbi  = (const float*)d_in[8];
  const float* gwhh = (const float*)d_in[9];
  const float* gbh  = (const float*)d_in[10];
  const float* hinit = (const float*)d_in[11];
  float* out = (float*)d_out;
  char* ws = (char*)d_ws;

  gru_prep_kernel<<<dim3(768), dim3(256), 0, stream>>>(W_w, gwih, gwhh, hinit, ws);
  gru_decoder_kernel<<<dim3(B_SZ), dim3(256), 0, stream>>>(
      signal, bases, msk, W_b, ln_g, ln_b, gbi, gbh, out, ws);
}
